// LinearScalarForest_52166672777550
// MI455X (gfx1250) — matrix operand order repaired and checked by reading, not compiled, not run on hardware
//
#include <hip/hip_runtime.h>
#include <hip/hip_bf16.h>

#ifndef USE_TDM
#define USE_TDM 1
#endif

// ---------------- problem constants ----------------
constexpr int B_  = 4096;
constexpr int D0_ = 128;
constexpr int D_  = 256;
constexpr int T_  = 64;
constexpr int NODES_ = 21;       // 1 + 4 + 16
constexpr int W_  = 4;
constexpr int S_  = 64;          // 4^3
constexpr int NW_ = NODES_ * W_; // 84 live logit columns per tree
constexpr int NP_ = 96;          // padded logit columns (6 WMMA tiles)
constexpr int BM_ = 16;          // batch rows per block (WMMA M)

// packed-weight workspace layout (bf16, K-contiguous per column)
constexpr size_t WLOG_ELEMS  = (size_t)T_ * NP_ * D_;   // 1.57M bf16 = 3.0 MB
constexpr size_t WPRED_ELEMS = (size_t)T_ * S_  * D_;   // 1.05M bf16 = 2.0 MB
constexpr size_t WS_NEEDED   = (WLOG_ELEMS + WPRED_ELEMS) * sizeof(unsigned short);

// ---------------- vector types for WMMA ----------------
typedef __bf16        v16bf __attribute__((ext_vector_type(16)));
typedef float         v8f   __attribute__((ext_vector_type(8)));
typedef unsigned int  u32x4 __attribute__((ext_vector_type(4)));
typedef int           i32x4 __attribute__((ext_vector_type(4)));
typedef int           i32x8 __attribute__((ext_vector_type(8)));

union Frag16 {                   // one WMMA 16-bit operand: 8 VGPRs / lane
    u32x4 q[2];
    v16bf v;
    unsigned short s[16];
};

__device__ __forceinline__ unsigned short f2bf(float f) {
    unsigned u = __builtin_bit_cast(unsigned, f);
    unsigned r = u + 0x7FFFu + ((u >> 16) & 1u);   // round-to-nearest-even
    return (unsigned short)(r >> 16);
}

#if USE_TDM && __has_builtin(__builtin_amdgcn_tensor_load_to_lds)
#define HAVE_TDM 1
// Build a 2D TDM descriptor per CDNA5 ISA §8 (D# group0/group1 bit layout)
// and issue TENSOR_LOAD_TO_LDS (6-arg builtin on this toolchain).
__device__ __forceinline__ void tdm_load_2d_f32(const void* gaddr, unsigned lds_off,
                                                unsigned width, unsigned rows,
                                                unsigned stride) {
    unsigned long long ga = (unsigned long long)(size_t)gaddr;
    u32x4 g0;
    g0[0] = 1u;                                      // count=1, user descriptor
    g0[1] = lds_off;                                 // lds_addr [63:32]
    g0[2] = (unsigned)(ga & 0xFFFFFFFFull);          // global_addr lo
    g0[3] = (unsigned)((ga >> 32) & 0x1FFFFFFull)    // global_addr hi (57b)
          | (2u << 30);                              // type=2 ("image")
    i32x8 g1;
    g1[0] = 0x00020000;                              // data_size=2 (4B), mask=0
    g1[1] = (int)(width << 16);                      // tensor_dim0[15:0] @63:48
    g1[2] = (int)((width >> 16) | (rows << 16));     // tensor_dim0 hi, tensor_dim1 lo
    g1[3] = (int)((rows >> 16) | (width << 16));     // tensor_dim1 hi, tile_dim0 @127:112
    g1[4] = (int)rows;                               // tile_dim1 @143:128 (tile_dim2=0)
    g1[5] = (int)stride;                             // tensor_dim0_stride lo32
    g1[6] = 0;
    g1[7] = 0;
    const i32x4 z4  = {0, 0, 0, 0};
    const i32x8 z8i = {0, 0, 0, 0, 0, 0, 0, 0};
    __builtin_amdgcn_tensor_load_to_lds(g0, g1, z4, z4, z8i, 0);
}
#else
#define HAVE_TDM 0
#endif

// ============ pack kernels: f32 weights -> bf16 [t][col][K] in d_ws ============
// Logits: dst[t][c][k] = Wn[t][c>>2][k][c&3] for c<84, else 0. One thread: 16 k's.
__global__ __launch_bounds__(256)
void pack_logit_weights(const float* __restrict__ Wn, unsigned short* __restrict__ dst) {
    const int idx = blockIdx.x * 256 + threadIdx.x;     // T*NP*(D/16) = 98304
    const int kq  = idx & 15;
    const int c   = (idx >> 4) % NP_;
    const int t   = idx / (16 * NP_);
    if (t >= T_) return;
    const int k0 = kq * 16;
    unsigned short v[16];
    if (c < NW_) {
        const float* src = Wn + ((size_t)(t * NODES_ + (c >> 2)) * D_ + k0) * W_ + (c & 3);
#pragma unroll
        for (int i = 0; i < 16; ++i) v[i] = f2bf(src[i * W_]);
    } else {
#pragma unroll
        for (int i = 0; i < 16; ++i) v[i] = 0;
    }
    u32x4 p0, p1;
#pragma unroll
    for (int q = 0; q < 4; ++q) {
        p0[q] = (unsigned)v[q*2]   | ((unsigned)v[q*2+1]   << 16);
        p1[q] = (unsigned)v[8+q*2] | ((unsigned)v[8+q*2+1] << 16);
    }
    u32x4* d = (u32x4*)(dst + ((size_t)(t * NP_ + c) * D_ + k0));
    d[0] = p0; d[1] = p1;
}

// Preds: dst[t][s][k] = (k<128 ? W0[t][s][k] : W1[t][s][k-128]). One thread: 16 k's.
__global__ __launch_bounds__(256)
void pack_pred_weights(const float* __restrict__ W0, const float* __restrict__ W1,
                       unsigned short* __restrict__ dst) {
    const int idx = blockIdx.x * 256 + threadIdx.x;     // T*S*(D/16) = 65536
    const int kq  = idx & 15;
    const int s   = (idx >> 4) & (S_ - 1);
    const int t   = idx / (16 * S_);
    if (t >= T_) return;
    const int k0 = kq * 16;
    const float* src = (k0 < D0_) ? (W0 + ((size_t)t * S_ + s) * D0_ + k0)
                                  : (W1 + ((size_t)t * S_ + s) * D0_ + (k0 - D0_));
    unsigned short v[16];
#pragma unroll
    for (int q = 0; q < 4; ++q) {
        float4 f = *(const float4*)(src + q * 4);
        v[q*4+0] = f2bf(f.x); v[q*4+1] = f2bf(f.y);
        v[q*4+2] = f2bf(f.z); v[q*4+3] = f2bf(f.w);
    }
    u32x4 p0, p1;
#pragma unroll
    for (int q = 0; q < 4; ++q) {
        p0[q] = (unsigned)v[q*2]   | ((unsigned)v[q*2+1]   << 16);
        p1[q] = (unsigned)v[8+q*2] | ((unsigned)v[8+q*2+1] << 16);
    }
    u32x4* d = (u32x4*)(dst + ((size_t)(t * S_ + s) * D_ + k0));
    d[0] = p0; d[1] = p1;
}

// =========================== fused forest kernel ===========================
template <bool PACKED>
__global__ __launch_bounds__(256)
void forest_fused(const float* __restrict__ inp0, const float* __restrict__ inp1,
                  const float* __restrict__ x2,
                  const float* __restrict__ Wn,  const float* __restrict__ bn,
                  const float* __restrict__ W0,  const float* __restrict__ b0v,
                  const float* __restrict__ W1,  const float* __restrict__ b1v,
                  const unsigned short* __restrict__ Wlogp,
                  const unsigned short* __restrict__ Wpredp,
                  float* __restrict__ out) {
    __shared__ __align__(16) unsigned short Xs[BM_][264];     // 8448 B
    __shared__ __align__(16) float Lp[8][BM_][88];            // 45056 B (also TDM stage)
    __shared__ float aveSh[BM_];

    const int tid    = threadIdx.x;
    const int b0     = blockIdx.x * BM_;
    const int wv     = tid >> 5;
    const int lane   = tid & 31;
    const int lane16 = lane & 15;
    const int hi     = lane >> 4;

    if (tid < BM_) aveSh[tid] = 0.f;

    // ---- stage X tile (16 x 256) as bf16 into LDS ----
    float* stage = &Lp[0][0][0];   // 16 KB f32 staging region, reused later as Lp
#if HAVE_TDM
    if (tid < 32) {                // wave 0 issues the two TDM loads
        unsigned soff = (unsigned)(unsigned long long)(size_t)(const void*)stage;
        tdm_load_2d_f32(inp0 + (size_t)b0 * D0_, soff,               D0_, BM_, D0_);
        tdm_load_2d_f32(inp1 + (size_t)b0 * D0_, soff + BM_*D0_*4,   D0_, BM_, D0_);
        __builtin_amdgcn_s_wait_tensorcnt(0);
    }
    __syncthreads();
#endif
    {
        const int row = tid >> 4;
        const int c0  = (tid & 15) * 16;
#if HAVE_TDM
        const float* src = (c0 < D0_) ? (stage + row * D0_ + c0)
                                      : (stage + BM_*D0_ + row * D0_ + (c0 - D0_));
#else
        const float* src = (c0 < D0_) ? (inp0 + (size_t)(b0 + row) * D0_ + c0)
                                      : (inp1 + (size_t)(b0 + row) * D0_ + (c0 - D0_));
#endif
        float fl[16];
#pragma unroll
        for (int q = 0; q < 4; ++q) {
            float4 f = *(const float4*)(src + q * 4);
            fl[q*4+0] = f.x; fl[q*4+1] = f.y; fl[q*4+2] = f.z; fl[q*4+3] = f.w;
        }
        u32x4 p0, p1;
#pragma unroll
        for (int q = 0; q < 4; ++q) {
            p0[q] = (unsigned)f2bf(fl[q*2])   | ((unsigned)f2bf(fl[q*2+1]) << 16);
            p1[q] = (unsigned)f2bf(fl[8+q*2]) | ((unsigned)f2bf(fl[8+q*2+1]) << 16);
        }
        u32x4* dst = (u32x4*)&Xs[row][c0];
        dst[0] = p0; dst[1] = p1;
    }
    __syncthreads();   // X tile ready; Lp region free for reuse

    const u32x4 zq = {0u, 0u, 0u, 0u};
    const v8f   z8 = {0.f, 0.f, 0.f, 0.f, 0.f, 0.f, 0.f, 0.f};
    float (*LpW)[88] = Lp[wv];

    float aves[8] = {0.f, 0.f, 0.f, 0.f, 0.f, 0.f, 0.f, 0.f};

    for (int tt = 0; tt < 8; ++tt) {
        const int t = wv * 8 + tt;
        if (t + 1 < T_) {   // warm L2 for next tree's weight blocks
            if constexpr (PACKED) {
                __builtin_prefetch(Wpredp + (size_t)(t + 1) * S_ * D_, 0, 1);
                __builtin_prefetch(Wlogp + (size_t)(t + 1) * NP_ * D_, 0, 1);
            } else {
                __builtin_prefetch(W0 + (size_t)(t + 1) * S_ * D0_, 0, 1);
                __builtin_prefetch(Wn + (size_t)(t + 1) * NODES_ * D_ * W_, 0, 1);
            }
        }

        // ================= GEMM1: node logits, 16 x 96 (84 live) =================
        v8f acc[6];
#pragma unroll
        for (int nt = 0; nt < 6; ++nt) acc[nt] = z8;

        for (int k0 = 0; k0 < D_; k0 += 32) {
            Frag16 A;   // 16x32 bf16 A fragment from LDS (contiguous k = 16*hi + s)
            A.q[0] = *(const u32x4*)&Xs[lane16][k0 + hi * 16];
            A.q[1] = *(const u32x4*)&Xs[lane16][k0 + hi * 16 + 8];
            const int dbase = k0 + hi * 16;
#pragma unroll
            for (int nt = 0; nt < 6; ++nt) {
                const int c = nt * 16 + lane16;     // column = node*4 + w
                Frag16 Bf;
                if constexpr (PACKED) {
                    const u32x4* wp =
                        (const u32x4*)(Wlogp + ((size_t)(t * NP_ + c) * D_ + dbase));
                    Bf.q[0] = wp[0]; Bf.q[1] = wp[1];
                } else {
                    if (c < NW_) {
                        const float* wp = Wn
                            + ((size_t)(t * NODES_ + (c >> 2)) * D_ + dbase) * W_ + (c & 3);
#pragma unroll
                        for (int i = 0; i < 16; ++i) Bf.s[i] = f2bf(wp[i * W_]);
                    } else {
                        Bf.q[0] = zq; Bf.q[1] = zq;
                    }
                }
                acc[nt] = __builtin_amdgcn_wmma_f32_16x16x32_bf16(
                    false, A.v, false, Bf.v, (short)0, acc[nt], false, false);
            }
        }

        // ---- softmax over each node's 4 children (adjacent lanes) -> probs to LDS
#pragma unroll
        for (int nt = 0; nt < 6; ++nt) {
            const int c = nt * 16 + lane16;
            const float bnode = (c < NW_)
                ? bn[(size_t)(t * NODES_ + (c >> 2)) * W_ + (c & 3)] : 0.f;
#pragma unroll
            for (int j = 0; j < 8; ++j) {
                float v  = acc[nt][j] + bnode;
                float mx = fmaxf(v, __shfl_xor(v, 1));
                mx       = fmaxf(mx, __shfl_xor(mx, 2));
                float e  = __expf(v - mx);
                float se = e + __shfl_xor(e, 1);
                se      += __shfl_xor(se, 2);
                float p  = e / se;
                if (c < NW_) LpW[j + 8 * hi][c] = p;
            }
        }

        // ================= GEMM2: leaf logits, 16 x 64 =================
        v8f acc2[4];
#pragma unroll
        for (int st = 0; st < 4; ++st) acc2[st] = z8;

        for (int k0 = 0; k0 < D_; k0 += 32) {
            Frag16 A;
            A.q[0] = *(const u32x4*)&Xs[lane16][k0 + hi * 16];
            A.q[1] = *(const u32x4*)&Xs[lane16][k0 + hi * 16 + 8];
            const int dbase = k0 + hi * 16;
#pragma unroll
            for (int st = 0; st < 4; ++st) {
                const int s = st * 16 + lane16;
                Frag16 Bf;
                if constexpr (PACKED) {
                    const u32x4* wp =
                        (const u32x4*)(Wpredp + ((size_t)(t * S_ + s) * D_ + dbase));
                    Bf.q[0] = wp[0]; Bf.q[1] = wp[1];
                } else {
                    const float* wp = (dbase < D0_)
                        ? (W0 + ((size_t)t * S_ + s) * D0_ + dbase)
                        : (W1 + ((size_t)t * S_ + s) * D0_ + (dbase - D0_));
#pragma unroll
                    for (int q = 0; q < 4; ++q) {
                        float4 f = *(const float4*)(wp + q * 4);
                        Bf.s[q*4+0] = f2bf(f.x); Bf.s[q*4+1] = f2bf(f.y);
                        Bf.s[q*4+2] = f2bf(f.z); Bf.s[q*4+3] = f2bf(f.w);
                    }
                }
                acc2[st] = __builtin_amdgcn_wmma_f32_16x16x32_bf16(
                    false, A.v, false, Bf.v, (short)0, acc2[st], false, false);
            }
        }

        // ---- epilogue: y = x2 + pred + b0 + b1 ; state = P0*P1*P2 ; accumulate
#pragma unroll
        for (int st = 0; st < 4; ++st) {
            const int s  = st * 16 + lane16;        // leaf index; w1 = st
            const int w2 = (s >> 2) & 3;
            const float bb = b0v[(size_t)t * S_ + s] + b1v[(size_t)t * S_ + s];
#pragma unroll
            for (int j = 0; j < 8; ++j) {
                const int r = j + 8 * hi;
                float y = acc2[st][j] + x2[b0 + r] + bb;
                out[((size_t)(b0 + r) * T_ + t) * S_ + s] = y;
                float p = LpW[r][st]                    // root prob, col w1
                        * LpW[r][4 + st * 4 + w2]       // level-1 prob
                        * LpW[r][20 + s];               // level-2 prob, col 20+s
                aves[j] += p * (1.f / (1.f + __expf(-y)));
            }
        }
    }

    // ---- block reduction of ave_pred over lanes/waves via LDS float atomics
#pragma unroll
    for (int j = 0; j < 8; ++j) atomicAdd(&aveSh[j + 8 * hi], aves[j]);
    __syncthreads();
    if (tid < BM_)
        out[(size_t)B_ * T_ * S_ + b0 + tid] = aveSh[tid] * (1.f / (float)T_);
}

extern "C" void kernel_launch(void* const* d_in, const int* in_sizes, int n_in,
                              void* d_out, int out_size, void* d_ws, size_t ws_size,
                              hipStream_t stream) {
    (void)in_sizes; (void)n_in; (void)out_size;
    const float* inp0 = (const float*)d_in[0];
    const float* inp1 = (const float*)d_in[1];
    const float* x2   = (const float*)d_in[2];
    const float* Wn   = (const float*)d_in[3];
    const float* bn   = (const float*)d_in[4];
    const float* W0   = (const float*)d_in[5];
    const float* b0   = (const float*)d_in[6];
    const float* W1   = (const float*)d_in[7];
    const float* b1   = (const float*)d_in[8];
    float* out = (float*)d_out;

    dim3 block(256);
    dim3 grid(B_ / BM_);

    if (d_ws != nullptr && ws_size >= WS_NEEDED) {
        unsigned short* Wlogp  = (unsigned short*)d_ws;
        unsigned short* Wpredp = Wlogp + WLOG_ELEMS;
        // one-pass weight repack (bf16, fragment-friendly layout) every call
        pack_logit_weights<<<dim3((T_ * NP_ * (D_ / 16)) / 256), block, 0, stream>>>(Wn, Wlogp);
        pack_pred_weights<<<dim3((T_ * S_ * (D_ / 16)) / 256), block, 0, stream>>>(W0, W1, Wpredp);
        forest_fused<true><<<grid, block, 0, stream>>>(
            inp0, inp1, x2, Wn, bn, W0, b0, W1, b1, Wlogp, Wpredp, out);
    } else {
        forest_fused<false><<<grid, block, 0, stream>>>(
            inp0, inp1, x2, Wn, bn, W0, b0, W1, b1, nullptr, nullptr, out);
    }
}
